// Attention_18597208392034
// MI455X (gfx1250) — matrix operand order repaired and checked by reading, not compiled, not run on hardware
//
#include <hip/hip_runtime.h>
#include <hip/hip_bf16.h>

// ---------------------------------------------------------------------------
// MHA block for MI455X (gfx1250): bf16 WMMA + async global->LDS double buffer.
//   B=8, N=1024, C=768, H=12, D=64, SCALE=1/8 (folded into Q at store time)
// ---------------------------------------------------------------------------

typedef __bf16 bf16_t;
typedef __bf16 v16bf  __attribute__((ext_vector_type(16)));
typedef __bf16 bf16x8 __attribute__((ext_vector_type(8)));
typedef float  v8f    __attribute__((ext_vector_type(8)));
typedef int    v4i    __attribute__((vector_size(16)));   // b128 payload type

#define WMMA_BF16(a, b, c) \
  __builtin_amdgcn_wmma_f32_16x16x32_bf16(false, (a), false, (b), (short)0, (c), false, false)

#if __has_builtin(__builtin_amdgcn_global_load_async_to_lds_b128) && \
    __has_builtin(__builtin_amdgcn_s_wait_asynccnt)
#define USE_ASYNC_LDS 1
#else
#define USE_ASYNC_LDS 0
#endif

// 16-byte global -> LDS copy. Async (ASYNCcnt-tracked, no VGPR data) when the
// gfx1250 builtins exist, else a synchronous VGPR round-trip.
static __device__ __forceinline__ void cp16_g2l(void* ldst, const void* gsrc) {
#if USE_ASYNC_LDS
  __builtin_amdgcn_global_load_async_to_lds_b128(
      (__attribute__((address_space(1))) v4i*)(unsigned long long)gsrc,
      (__attribute__((address_space(3))) v4i*)(unsigned)(unsigned long long)ldst,
      0, 0);
#else
  *(bf16x8*)ldst = *(const bf16x8*)gsrc;
#endif
}
static __device__ __forceinline__ void async_wait_keep4() {
#if USE_ASYNC_LDS
  __builtin_amdgcn_s_wait_asynccnt(4);   // allow the 4 newest (next tile) to fly
#endif
}
static __device__ __forceinline__ void async_wait_all() {
#if USE_ASYNC_LDS
  __builtin_amdgcn_s_wait_asynccnt(0);
#endif
}

static __device__ __forceinline__ v16bf make_v16(bf16x8 lo, bf16x8 hi8) {
  v16bf a;
#pragma unroll
  for (int i = 0; i < 8; ++i) { a[i] = lo[i]; a[i + 8] = hi8[i]; }
  return a;
}

// ---------------------------------------------------------------------------
// Stage 0: fp32 -> bf16 conversions (elementwise + transposed weights)
// ---------------------------------------------------------------------------
__global__ void cvt_f32_to_bf16(const float* __restrict__ in, bf16_t* __restrict__ out, int n) {
  int i = blockIdx.x * 256 + threadIdx.x;
  if (i < n) out[i] = (bf16_t)in[i];
}

__global__ void transpose_f32_to_bf16(const float* __restrict__ in, bf16_t* __restrict__ out,
                                      int rows, int cols) {
  int i = blockIdx.x * 256 + threadIdx.x;
  if (i < rows * cols) {
    int r = i / cols, c = i % cols;
    out[(size_t)c * rows + r] = (bf16_t)in[i];
  }
}

// ---------------------------------------------------------------------------
// Shared GEMM core: block = 4 waves, 64x64 output tile, K in steps of 32.
// Per step, 64x32 A and B tiles (4KB each) staged into LDS via async copies,
// double buffered (16KB). Wave w computes rows w*16..w*16+15 (4 WMMAs/step).
// Last iteration is peeled so the loop body is branch-free (avoids acc-phi
// register copies); all fragments are loaded before the WMMA chain so the ds
// loads clause together behind a single dscnt wait.
//   smem + bi*4096        : A tile (64 rows x 64B)
//   smem + 8192 + bi*4096 : B tile (64 rows x 64B)
// ---------------------------------------------------------------------------
static __device__ __forceinline__ void gemm_core_64x64(
    const bf16_t* __restrict__ Arows,   // already offset to row m0, stride K
    const bf16_t* __restrict__ Brows,   // already offset to row n0, stride K
    int K, char* smem /* 16KB */,
    v8f& acc0, v8f& acc1, v8f& acc2, v8f& acc3) {
  const int tid  = threadIdx.x;
  const int lane = tid & 31;
  const int wave = tid >> 5;
  const int row  = lane & 15;
  const int half = lane >> 4;

  const int crow = tid >> 1;            // 0..63 : tile row this thread copies
  const int cofs = (tid & 1) * 32;      // byte offset within 64B LDS row
  const int gofs = (tid & 1) * 16;      // element offset within 32-wide k chunk

  auto issue = [&](int kk, int bi) {
    char* Ab = smem + bi * 4096;
    char* Bb = smem + 8192 + bi * 4096;
    const bf16_t* ga = Arows + (size_t)crow * K + kk + gofs;
    cp16_g2l(Ab + crow * 64 + cofs,      ga);
    cp16_g2l(Ab + crow * 64 + cofs + 16, ga + 8);
    const bf16_t* gb = Brows + (size_t)crow * K + kk + gofs;
    cp16_g2l(Bb + crow * 64 + cofs,      gb);
    cp16_g2l(Bb + crow * 64 + cofs + 16, gb + 8);
  };

  auto compute = [&](int bi) {
    const char* ar = smem + bi * 4096 + (wave * 16 + row) * 64;
    const char* bb = smem + 8192 + bi * 4096 + row * 64 + half * 32;
    // Load A-frag + all four B-frags first, then run the WMMA chain.
    const v16bf a  = make_v16(*(const bf16x8*)(ar + half * 32),
                              *(const bf16x8*)(ar + 16 + half * 32));
    const v16bf b0 = *(const v16bf*)(bb);
    const v16bf b1 = *(const v16bf*)(bb + 16 * 64);
    const v16bf b2 = *(const v16bf*)(bb + 32 * 64);
    const v16bf b3 = *(const v16bf*)(bb + 48 * 64);
    acc0 = WMMA_BF16(a, b0, acc0);
    acc1 = WMMA_BF16(a, b1, acc1);
    acc2 = WMMA_BF16(a, b2, acc2);
    acc3 = WMMA_BF16(a, b3, acc3);
  };

  const int NK = K / 32;
  issue(0, 0);
  for (int i = 0; i < NK - 1; ++i) {          // branch-free steady state
    const int bi = i & 1;
    issue((i + 1) * 32, bi ^ 1);
    async_wait_keep4();
    __syncthreads();                          // tile i visible to all waves
    compute(bi);
    __syncthreads();                          // done reading buf bi
  }
  async_wait_all();                           // peeled final iteration
  __syncthreads();
  compute((NK - 1) & 1);
}

// ---------------------------------------------------------------------------
// Stage 1: QKV GEMM -> scatter to Q (pre-scaled), K [b,h,n,d], V^T [b,h,d,n]
// ---------------------------------------------------------------------------
__global__ __launch_bounds__(128) void qkv_gemm_bf16(
    const bf16_t* __restrict__ X, const bf16_t* __restrict__ Wt,
    bf16_t* __restrict__ Qb, bf16_t* __restrict__ Kb, bf16_t* __restrict__ Vt) {
  __shared__ char smem[16384];
  const int lane = threadIdx.x & 31;
  const int wave = threadIdx.x >> 5;
  const int row  = lane & 15;
  const int half = lane >> 4;
  const int n0   = blockIdx.x * 64;
  const int m0   = blockIdx.y * 64;

  v8f acc0 = {}, acc1 = {}, acc2 = {}, acc3 = {};
  gemm_core_64x64(X + (size_t)m0 * 768, Wt + (size_t)n0 * 768, 768, smem,
                  acc0, acc1, acc2, acc3);
  const v8f acc[4] = { acc0, acc1, acc2, acc3 };

#pragma unroll
  for (int s = 0; s < 4; ++s) {
    const int c    = n0 + s * 16 + row;
    const int sect = c / 768;
    const int rem  = c % 768;
    const int h    = rem >> 6;
    const int d    = rem & 63;
#pragma unroll
    for (int r = 0; r < 8; ++r) {
      const int m  = m0 + wave * 16 + r + half * 8;
      const int bb = m >> 10;
      const int nr = m & 1023;
      const float v = acc[s][r];
      if (sect == 0)
        Qb[((size_t)(bb * 12 + h) * 1024 + nr) * 64 + d] = (bf16_t)(v * 0.125f);
      else if (sect == 1)
        Kb[((size_t)(bb * 12 + h) * 1024 + nr) * 64 + d] = (bf16_t)v;
      else
        Vt[((size_t)(bb * 12 + h) * 64 + d) * 1024 + nr] = (bf16_t)v;
    }
  }
}

// ---------------------------------------------------------------------------
// Stage 2: flash attention. K/V tiles (4KB each) shared by the 4 waves via
// async double buffer; online softmax in C-tile layout; P transposed to an
// A-fragment through a per-wave LDS tile (16x32 bf16, 80B row stride).
// LDS: smem + bi*4096        : K tile (32 keys x 128B)
//      smem + 8192 + bi*4096 : V tile (64 d x 64B)
//      smem + 16384 + w*1280 : per-wave P staging
// ---------------------------------------------------------------------------
__global__ __launch_bounds__(128) void flash_attn_bf16(
    const bf16_t* __restrict__ Q, const bf16_t* __restrict__ Kc,
    const bf16_t* __restrict__ Vt, bf16_t* __restrict__ O) {
  __shared__ char smem[2 * 4096 + 2 * 4096 + 4 * 16 * 80];
  const int tid  = threadIdx.x;
  const int lane = tid & 31;
  const int wave = tid >> 5;
  const int row  = lane & 15;
  const int half = lane >> 4;
  const int h    = blockIdx.y;
  const int b    = blockIdx.z;
  const int t0   = blockIdx.x * 64 + wave * 16;
  const int bh   = b * 12 + h;

  const bf16_t* Kbh = Kc + (size_t)bh * (1024 * 64);
  const bf16_t* Vbh = Vt + (size_t)bh * (64 * 1024);

  // Preload Q A-fragments (16 rows x 64 d => two k=32 fragments)
  const bf16_t* qrow = Q + ((size_t)bh * 1024 + (t0 + row)) * 64;
  const v16bf qa0 = make_v16(*(const bf16x8*)(qrow +      half * 16),
                             *(const bf16x8*)(qrow + 8 + half * 16));
  const v16bf qa1 = make_v16(*(const bf16x8*)(qrow + 32 + half * 16),
                             *(const bf16x8*)(qrow + 40 + half * 16));

  auto issue = [&](int s0, int bi) {
    {  // K tile: row = key (32), 128B row
      char* Kt = smem + bi * 4096;
      const int kr = tid >> 2;
      const int co = (tid & 3) * 32;
      const bf16_t* g = Kbh + (size_t)(s0 + kr) * 64 + (tid & 3) * 16;
      cp16_g2l(Kt + kr * 128 + co,      g);
      cp16_g2l(Kt + kr * 128 + co + 16, g + 8);
    }
    {  // V tile: row = d (64), 64B row
      char* Vb = smem + 8192 + bi * 4096;
      const int vr = tid >> 1;
      const int co = (tid & 1) * 32;
      const bf16_t* g = Vbh + (size_t)vr * 1024 + s0 + (tid & 1) * 16;
      cp16_g2l(Vb + vr * 64 + co,      g);
      cp16_g2l(Vb + vr * 64 + co + 16, g + 8);
    }
  };

  v8f o0 = {}, o1 = {}, o2 = {}, o3 = {};
  float mrow[8], lrow[8];
#pragma unroll
  for (int r = 0; r < 8; ++r) { mrow[r] = -1e30f; lrow[r] = 0.f; }

  auto step = [&](int bi) {
    const char* Kt   = smem + bi * 4096;
    const char* Vb   = smem + 8192 + bi * 4096;
    char*       Pbuf = smem + 16384 + wave * (16 * 80);

    // ---- scores: load all K B-frags, then the 4-WMMA chain ----
    const char* kp0 = Kt + row * 128 + half * 32;
    const char* kp1 = Kt + (16 + row) * 128 + half * 32;
    const v16bf k00 = *(const v16bf*)(kp0);
    const v16bf k01 = *(const v16bf*)(kp0 + 64);
    const v16bf k10 = *(const v16bf*)(kp1);
    const v16bf k11 = *(const v16bf*)(kp1 + 64);
    v8f S0 = {}, S1 = {};
    S0 = WMMA_BF16(qa0, k00, S0);
    S0 = WMMA_BF16(qa1, k01, S0);
    S1 = WMMA_BF16(qa0, k10, S1);
    S1 = WMMA_BF16(qa1, k11, S1);

    // ---- online softmax in C-tile layout ----
#pragma unroll
    for (int r = 0; r < 8; ++r) {
      float t = fmaxf(S0[r], S1[r]);
      t = fmaxf(t, __shfl_xor(t, 1, 32));
      t = fmaxf(t, __shfl_xor(t, 2, 32));
      t = fmaxf(t, __shfl_xor(t, 4, 32));
      t = fmaxf(t, __shfl_xor(t, 8, 32));
      const float mn = fmaxf(mrow[r], t);
      const float sc = __expf(mrow[r] - mn);
      mrow[r] = mn;
      const float p0 = __expf(S0[r] - mn);
      const float p1 = __expf(S1[r] - mn);
      float rs = p0 + p1;
      rs += __shfl_xor(rs, 1, 32);
      rs += __shfl_xor(rs, 2, 32);
      rs += __shfl_xor(rs, 4, 32);
      rs += __shfl_xor(rs, 8, 32);
      lrow[r] = lrow[r] * sc + rs;
      o0[r] *= sc; o1[r] *= sc; o2[r] *= sc; o3[r] *= sc;
      bf16_t* prow = (bf16_t*)(Pbuf + (r + half * 8) * 80);
      prow[row]      = (bf16_t)p0;
      prow[row + 16] = (bf16_t)p1;
    }
    asm volatile("" ::: "memory");   // wave-local LDS staging, ds ops in-order

    // ---- reload P as 16x32 A-fragment; O += P @ V (frags loaded upfront) ----
    const char* myrow = Pbuf + row * 80;
    const v16bf pa = make_v16(*(const bf16x8*)(myrow + half * 32),
                              *(const bf16x8*)(myrow + 16 + half * 32));
    const char* vp = Vb + row * 64 + half * 32;
    const v16bf v0 = *(const v16bf*)(vp);
    const v16bf v1 = *(const v16bf*)(vp + 16 * 64);
    const v16bf v2 = *(const v16bf*)(vp + 32 * 64);
    const v16bf v3 = *(const v16bf*)(vp + 48 * 64);
    o0 = WMMA_BF16(pa, v0, o0);
    o1 = WMMA_BF16(pa, v1, o1);
    o2 = WMMA_BF16(pa, v2, o2);
    o3 = WMMA_BF16(pa, v3, o3);
  };

  issue(0, 0);
  for (int i = 0; i < 31; ++i) {              // branch-free steady state
    const int bi = i & 1;
    issue((i + 1) * 32, bi ^ 1);
    async_wait_keep4();
    __syncthreads();
    step(bi);
    __syncthreads();
  }
  async_wait_all();                           // peeled final iteration
  __syncthreads();
  step(1);

  // ---- normalize and write [b, t, h*64 + d] as bf16 ----
  const v8f oo[4] = { o0, o1, o2, o3 };
#pragma unroll
  for (int dt = 0; dt < 4; ++dt) {
#pragma unroll
    for (int r = 0; r < 8; ++r) {
      const int t = t0 + r + half * 8;
      const float v = oo[dt][r] / lrow[r];
      O[((size_t)b * 1024 + t) * 768 + h * 64 + dt * 16 + row] = (bf16_t)v;
    }
  }
}

// ---------------------------------------------------------------------------
// Stage 3: projection GEMM + bias, f32 output.
// ---------------------------------------------------------------------------
__global__ __launch_bounds__(128) void proj_gemm_bf16(
    const bf16_t* __restrict__ A, const bf16_t* __restrict__ Wt,
    const float* __restrict__ bias, float* __restrict__ out) {
  __shared__ char smem[16384];
  const int lane = threadIdx.x & 31;
  const int wave = threadIdx.x >> 5;
  const int row  = lane & 15;
  const int half = lane >> 4;
  const int n0   = blockIdx.x * 64;
  const int m0   = blockIdx.y * 64;

  v8f acc0 = {}, acc1 = {}, acc2 = {}, acc3 = {};
  gemm_core_64x64(A + (size_t)m0 * 768, Wt + (size_t)n0 * 768, 768, smem,
                  acc0, acc1, acc2, acc3);
  const v8f acc[4] = { acc0, acc1, acc2, acc3 };

#pragma unroll
  for (int s = 0; s < 4; ++s) {
    const int c = n0 + s * 16 + row;
    const float bb = bias[c];
#pragma unroll
    for (int r = 0; r < 8; ++r) {
      const int m = m0 + wave * 16 + r + half * 8;
      out[(size_t)m * 768 + c] = acc[s][r] + bb;
    }
  }
}

// ---------------------------------------------------------------------------
// Launcher.  Workspace: Xbf, WqkvT, WprojT, Q, K, Vt, attnO  => ~67.6 MB
// ---------------------------------------------------------------------------
extern "C" void kernel_launch(void* const* d_in, const int* in_sizes, int n_in,
                              void* d_out, int out_size, void* d_ws, size_t ws_size,
                              hipStream_t stream) {
  (void)in_sizes; (void)n_in; (void)out_size; (void)ws_size;
  const float* x     = (const float*)d_in[0];
  const float* Wqkv  = (const float*)d_in[1];
  const float* Wproj = (const float*)d_in[2];
  const float* bproj = (const float*)d_in[3];
  float* out = (float*)d_out;

  char* ws = (char*)d_ws;
  const size_t SZ_X   = (size_t)8192 * 768 * 2;
  const size_t SZ_WQ  = (size_t)2304 * 768 * 2;
  const size_t SZ_WP  = (size_t)768 * 768 * 2;
  const size_t SZ_QKV = (size_t)8 * 12 * 1024 * 64 * 2;

  bf16_t* Xbf    = (bf16_t*)(ws);
  bf16_t* WqkvT  = (bf16_t*)(ws + SZ_X);
  bf16_t* WprojT = (bf16_t*)(ws + SZ_X + SZ_WQ);
  bf16_t* Qb     = (bf16_t*)(ws + SZ_X + SZ_WQ + SZ_WP);
  bf16_t* Kb     = (bf16_t*)(ws + SZ_X + SZ_WQ + SZ_WP + SZ_QKV);
  bf16_t* Vt     = (bf16_t*)(ws + SZ_X + SZ_WQ + SZ_WP + 2 * SZ_QKV);
  bf16_t* Obf    = (bf16_t*)(ws + SZ_X + SZ_WQ + SZ_WP + 3 * SZ_QKV);

  cvt_f32_to_bf16<<<(8192 * 768 + 255) / 256, 256, 0, stream>>>(x, Xbf, 8192 * 768);
  transpose_f32_to_bf16<<<(768 * 2304 + 255) / 256, 256, 0, stream>>>(Wqkv, WqkvT, 768, 2304);
  transpose_f32_to_bf16<<<(768 * 768 + 255) / 256, 256, 0, stream>>>(Wproj, WprojT, 768, 768);

  qkv_gemm_bf16<<<dim3(36, 128), 128, 0, stream>>>(Xbf, WqkvT, Qb, Kb, Vt);
  flash_attn_bf16<<<dim3(16, 12, 8), 128, 0, stream>>>(Qb, Kb, Vt, Obf);
  proj_gemm_bf16<<<dim3(12, 128), 128, 0, stream>>>(Obf, WprojT, bproj, out);
}
